// HybridKernel_65481071409204
// MI455X (gfx1250) — matrix operand order repaired and checked by reading, not compiled, not run on hardware
//
#include <hip/hip_runtime.h>

typedef __attribute__((ext_vector_type(16))) __bf16 v16bf;
typedef __attribute__((ext_vector_type(8)))  __bf16 v8bf;
typedef __attribute__((ext_vector_type(8)))  float  v8f;

#define N_ROWS 8192
#define K_DIM  256
#define TILE   128
#define LDSS   40   // padded LDS row stride in bf16 elements (32 -> 40, kills bank conflicts)

__device__ __forceinline__ v16bf concat16(v8bf a, v8bf b) {
    return __builtin_shufflevector(a, b, 0,1,2,3,4,5,6,7,8,9,10,11,12,13,14,15);
}

// ------------------------------------------------------------------
// Kernel 1: row norms  ws[0..8191] = ||x_i||^2, ws[8192..16383] = ||y_j||^2
// One wave32 per row; lane-coalesced loads + shuffle reduction.
// ------------------------------------------------------------------
__global__ __launch_bounds__(256) void rbfpoly_norms(const float* __restrict__ x,
                                                     const float* __restrict__ y,
                                                     float* __restrict__ ws) {
    int gwave = (int)(blockIdx.x * blockDim.x + threadIdx.x) >> 5;
    int lane  = threadIdx.x & 31;
    const float* src;
    float* dst;
    int row;
    if (gwave < N_ROWS) { src = x; dst = ws;          row = gwave; }
    else                { src = y; dst = ws + N_ROWS; row = gwave - N_ROWS; }
    const float* p = src + (size_t)row * K_DIM;
    float s = 0.f;
    #pragma unroll
    for (int k = 0; k < K_DIM; k += 32) {
        float v = p[k + lane];
        s = fmaf(v, v, s);
    }
    #pragma unroll
    for (int off = 16; off > 0; off >>= 1)
        s += __shfl_down(s, off, 32);
    if (lane == 0) dst[row] = s;
}

// ------------------------------------------------------------------
// Kernel 2: fused bf16x3-split WMMA GEMM + RBF/poly epilogue.
// 128x128 tile per 256-thread block; 8 waves in a 2(row) x 4(col) grid,
// each wave owns a 64x32 sub-tile = 4x2 accumulators of 16x16 f32.
// Double-buffered LDS (ping-pong): stage chunk i+1 while WMMAs of chunk i
// run on the XDL pipe; one barrier per iteration.
// ------------------------------------------------------------------
__global__ __launch_bounds__(256) void rbfpoly_gemm(const float* __restrict__ x,
                                                    const float* __restrict__ y,
                                                    const float* __restrict__ ws,
                                                    float* __restrict__ out) {
    __shared__ __bf16 Ahi[2][TILE * LDSS];
    __shared__ __bf16 Alo[2][TILE * LDSS];
    __shared__ __bf16 Bhi[2][TILE * LDSS];
    __shared__ __bf16 Blo[2][TILE * LDSS];

    const int tid  = threadIdx.x;
    const int lane = tid & 31;
    const int wave = tid >> 5;
    const int wm   = wave & 1;   // wave-row: 2 x 64 rows
    const int wn   = wave >> 1;  // wave-col: 4 x 32 cols

    const int rowBlk = blockIdx.y * TILE;
    const int colBlk = blockIdx.x * TILE;

    // global->LDS staging: 2 threads per tile row, each covers 16 of 32 K values
    const int ldRow   = tid >> 1;
    const int ldHalf  = tid & 1;
    const float* gx   = x + (size_t)(rowBlk + ldRow) * K_DIM + ldHalf * 16;
    const float* gy   = y + (size_t)(colBlk + ldRow) * K_DIM + ldHalf * 16;
    const int ldsBase = ldRow * LDSS + ldHalf * 16;

    v8f acc[4][2];
    #pragma unroll
    for (int m = 0; m < 4; ++m)
        #pragma unroll
        for (int n = 0; n < 2; ++n)
            #pragma unroll
            for (int e = 0; e < 8; ++e)
                acc[m][n][e] = 0.f;

    float4 xa[4], ya[4];   // staging registers for the in-flight chunk

    auto loadNext = [&](int kc) {
        #pragma unroll
        for (int q = 0; q < 4; ++q) {
            xa[q] = *(const float4*)(gx + kc + 4 * q);
            ya[q] = *(const float4*)(gy + kc + 4 * q);
        }
    };

    auto stageTo = [&](int b) {
        float fx[16] = {xa[0].x, xa[0].y, xa[0].z, xa[0].w,
                        xa[1].x, xa[1].y, xa[1].z, xa[1].w,
                        xa[2].x, xa[2].y, xa[2].z, xa[2].w,
                        xa[3].x, xa[3].y, xa[3].z, xa[3].w};
        float fy[16] = {ya[0].x, ya[0].y, ya[0].z, ya[0].w,
                        ya[1].x, ya[1].y, ya[1].z, ya[1].w,
                        ya[2].x, ya[2].y, ya[2].z, ya[2].w,
                        ya[3].x, ya[3].y, ya[3].z, ya[3].w};
        v8bf xh0, xh1, xl0, xl1, yh0, yh1, yl0, yl1;
        #pragma unroll
        for (int i = 0; i < 8; ++i) {
            __bf16 a = (__bf16)fx[i];
            xh0[i] = a;  xl0[i] = (__bf16)(fx[i] - (float)a);
            __bf16 b2 = (__bf16)fx[8 + i];
            xh1[i] = b2; xl1[i] = (__bf16)(fx[8 + i] - (float)b2);
            __bf16 c = (__bf16)fy[i];
            yh0[i] = c;  yl0[i] = (__bf16)(fy[i] - (float)c);
            __bf16 d = (__bf16)fy[8 + i];
            yh1[i] = d;  yl1[i] = (__bf16)(fy[8 + i] - (float)d);
        }
        *(v8bf*)&Ahi[b][ldsBase]     = xh0;
        *(v8bf*)&Ahi[b][ldsBase + 8] = xh1;
        *(v8bf*)&Alo[b][ldsBase]     = xl0;
        *(v8bf*)&Alo[b][ldsBase + 8] = xl1;
        *(v8bf*)&Bhi[b][ldsBase]     = yh0;
        *(v8bf*)&Bhi[b][ldsBase + 8] = yh1;
        *(v8bf*)&Blo[b][ldsBase]     = yl0;
        *(v8bf*)&Blo[b][ldsBase + 8] = yl1;
    };

    const int am = lane & 15;
    const int ak = (lane >> 4) * 8;
    const int bk = (lane >> 4) * 8;

    auto compute = [&](int b) {
        // A fragments: lanes 0-15 row M=lane K{0..7,16..23}; lanes 16-31 same M, K{8..15,24..31}
        v16bf ah[4], al[4];
        #pragma unroll
        for (int m = 0; m < 4; ++m) {
            int base = (wm * 64 + m * 16 + am) * LDSS + ak;
            ah[m] = concat16(*(const v8bf*)&Ahi[b][base], *(const v8bf*)&Ahi[b][base + 16]);
            al[m] = concat16(*(const v8bf*)&Alo[b][base], *(const v8bf*)&Alo[b][base + 16]);
        }
        // B fragments: lanes 0-15 col N=lane K{0..7,16..23}; lanes 16-31 col N=lane-16 K{8..15,24..31}
        v16bf bh[2], bl[2];
        #pragma unroll
        for (int n = 0; n < 2; ++n) {
            int base = (wn * 32 + n * 16 + am) * LDSS + bk;
            bh[n] = concat16(*(const v8bf*)&Bhi[b][base], *(const v8bf*)&Bhi[b][base + 16]);
            bl[n] = concat16(*(const v8bf*)&Blo[b][base], *(const v8bf*)&Blo[b][base + 16]);
        }
        // bf16x3 split accumulation: hi*hi + hi*lo + lo*hi (fp32 acc)
        #pragma unroll
        for (int m = 0; m < 4; ++m)
            #pragma unroll
            for (int n = 0; n < 2; ++n) {
                acc[m][n] = __builtin_amdgcn_wmma_f32_16x16x32_bf16(
                    false, ah[m], false, bh[n], (short)0, acc[m][n], false, false);
                acc[m][n] = __builtin_amdgcn_wmma_f32_16x16x32_bf16(
                    false, ah[m], false, bl[n], (short)0, acc[m][n], false, false);
                acc[m][n] = __builtin_amdgcn_wmma_f32_16x16x32_bf16(
                    false, al[m], false, bh[n], (short)0, acc[m][n], false, false);
            }
    };

    // ---- software pipeline: stage chunk 0, then ping-pong ----
    loadNext(0);
    stageTo(0);
    #pragma unroll 2
    for (int i = 0; i < K_DIM / 32; ++i) {
        __syncthreads();                         // buf[i&1] visible; buf[(i+1)&1] free (WAR)
        if (i < K_DIM / 32 - 1) loadNext(32 * (i + 1));   // issue global loads early
        compute(i & 1);                          // 24 WMMAs on XDL pipe
        if (i < K_DIM / 32 - 1) stageTo((i + 1) & 1);     // converts/stores co-execute with WMMA
    }

    // ---- fused epilogue: sqdist -> exp + poly, straight to global ----
    const float* x2 = ws;
    const float* y2 = ws + N_ROWS;
    const int colLane = lane & 15;
    const int rowSel  = (lane >> 4) * 8;   // C layout: VGPR e -> M=e (lanes 0-15) / M=e+8 (16-31)
    #pragma unroll
    for (int m = 0; m < 4; ++m) {
        const int gr0 = rowBlk + wm * 64 + m * 16 + rowSel;
        #pragma unroll
        for (int n = 0; n < 2; ++n) {
            const int gc  = colBlk + wn * 32 + n * 16 + colLane;
            const float y2v = y2[gc];
            #pragma unroll
            for (int e = 0; e < 8; ++e) {
                const int gr = gr0 + e;
                const float xy = acc[m][n][e];
                const float sq = fmaxf(x2[gr] + y2v - 2.0f * xy, 0.0f);
                const float t  = xy + 1.0f;
                out[(size_t)gr * N_ROWS + gc] = __expf(-0.5f * sq) + 0.1f * (t * t * t);
            }
        }
    }
}

extern "C" void kernel_launch(void* const* d_in, const int* in_sizes, int n_in,
                              void* d_out, int out_size, void* d_ws, size_t ws_size,
                              hipStream_t stream) {
    const float* x = (const float*)d_in[0];
    const float* y = (const float*)d_in[1];
    float* out = (float*)d_out;
    float* ws  = (float*)d_ws;   // 2*8192 floats of row norms

    // 2*8192 rows, one wave32 each, 8 waves per block -> 2048 blocks
    rbfpoly_norms<<<(2 * N_ROWS) / 8, 256, 0, stream>>>(x, y, ws);

    dim3 grid(N_ROWS / TILE, N_ROWS / TILE);  // 64 x 64
    rbfpoly_gemm<<<grid, 256, 0, stream>>>(x, y, ws, out);
}
